// GGL_70987219468903
// MI455X (gfx1250) — matrix operand order repaired and checked by reading, not compiled, not run on hardware
//
#include <hip/hip_runtime.h>
#include <hip/hip_bf16.h>

typedef __attribute__((ext_vector_type(16))) _Float16 v16h;
typedef __attribute__((ext_vector_type(8)))  float    v8f;

#define B_SZ   8
#define N_SZ   4096
#define D_SZ   128
#define KF     10          // projection features
#define KPAD   32          // padded feature storage (halves per node, upper 22 zero)
#define TOPK   32
#define ROWS_PER_BLK 16    // one 16-row WMMA block per workgroup
#define WAVES  8           // 256 threads

// ---------------------------------------------------------------------------
// Kernel 1: xt = leaky_relu(x @ W^T + b), stored as f16 padded to 32 features
// (K=10 real, K=10..31 zero). The zero tail lets WMMA B-fragment hi lanes
// (K=16..31) load real zeros instead of masking with v_pk_mul_f16.
// ---------------------------------------------------------------------------
__global__ void __launch_bounds__(256) xt_kernel(const float* __restrict__ x,
                                                 const float* __restrict__ W,
                                                 const float* __restrict__ bias,
                                                 _Float16* __restrict__ xt) {
    int gid = blockIdx.x * blockDim.x + threadIdx.x;   // 0 .. B*N-1
    const float4* xr = (const float4*)(x + (size_t)gid * D_SZ);
    const float4* W4 = (const float4*)W;

    float acc[KF];
#pragma unroll
    for (int k = 0; k < KF; ++k) acc[k] = bias[k];

    for (int d = 0; d < D_SZ / 4; ++d) {
        float4 xv = xr[d];
#pragma unroll
        for (int k = 0; k < KF; ++k) {
            float4 wv = W4[k * (D_SZ / 4) + d];
            acc[k] += xv.x * wv.x + xv.y * wv.y + xv.z * wv.z + xv.w * wv.w;
        }
    }

    v16h h, z;
#pragma unroll
    for (int k = 0; k < 16; ++k) { h[k] = (_Float16)0.f; z[k] = (_Float16)0.f; }
#pragma unroll
    for (int k = 0; k < KF; ++k) {
        float v = acc[k];
        v = (v > 0.f) ? v : 0.01f * v;                 // leaky_relu, slope 0.01
        h[k] = (_Float16)v;
    }
    _Float16* row = xt + (size_t)gid * KPAD;           // 64B-aligned row
    *(v16h*)(row)      = h;                            // K = 0..15  (10 real)
    *(v16h*)(row + 16) = z;                            // K = 16..31 (all zero)
}

// ---------------------------------------------------------------------------
// Kernel 2: one workgroup per (b, 16-row block).
//   Phase 1: WMMA Gram tiles (16 x 4096 slice of sim) -> LDS, scaled + diag mask
//   Phase 2: per-row top-32; persistent per-lane local max, owner-only rescan
//   Phase 3: sigmoid(bond) write + prefetch + gather of x rows
// ---------------------------------------------------------------------------
__global__ void __launch_bounds__(256, 1) topk_kernel(const float* __restrict__ x,
                                                      const _Float16* __restrict__ xt,
                                                      const float* __restrict__ ew,
                                                      float* __restrict__ out_nn,
                                                      float* __restrict__ out_bn) {
    extern __shared__ float lds[];
    float* sim  = lds;                                  // [16][4096] = 256 KB
    float* tval = lds + ROWS_PER_BLK * N_SZ;            // [16*32]
    int*   tidx = (int*)(tval + ROWS_PER_BLK * TOPK);   // [16*32]

    const int blk    = blockIdx.x;                      // 0 .. 2047
    const int b      = blk >> 8;                        // / (N/16)
    const int nbase  = (blk & 255) * ROWS_PER_BLK;
    const int tid    = threadIdx.x;
    const int wave   = tid >> 5;
    const int lane   = tid & 31;
    const int lane16 = lane & 15;
    const bool hi    = lane >= 16;

    const float NEG_INF = -__builtin_huge_valf();
    const float scale   = __expf(ew[0]);

    // ---- A fragment: 16x32 f16, M = lane&15.
    // lanes 0-15 hold K=0..15; lanes 16-31 hold K=16..31 (zero padded).
    const _Float16* arow = xt + ((size_t)(b * N_SZ + nbase + lane16)) * KPAD;
    v16h afrag;
#pragma unroll
    for (int j = 0; j < 16; ++j) afrag[j] = arow[(hi ? 16 : 0) + j];

    // ---- Phase 1: each wave owns 32 column tiles (16 cols each)
    for (int t = 0; t < 32; ++t) {
        int ct  = wave * 32 + t;
        int col = ct * 16 + lane16;

        // B fragment: 32x16 f16, N = lane&15. lanes 0-15 carry K=0..15,
        // lanes 16-31 carry K=16..31 -> load the zero-padded tail (real zeros).
        v16h bfrag = *(const v16h*)(xt + ((size_t)(b * N_SZ + col)) * KPAD
                                       + (hi ? 16 : 0));

        v8f c = {0.f, 0.f, 0.f, 0.f, 0.f, 0.f, 0.f, 0.f};
        c = __builtin_amdgcn_wmma_f32_16x16x32_f16(false, afrag, false, bfrag,
                                                   (short)0, c, false, false);

        // C layout: VGPR i -> M = i (lanes 0-15) / 8+i (lanes 16-31); N = lane&15
#pragma unroll
        for (int i = 0; i < 8; ++i) {
            int M  = i + (hi ? 8 : 0);
            int gr = nbase + M;
            int gc = ct * 16 + lane16;
            float v = c[i] * scale;
            if (gr == gc) v = NEG_INF;                  // mask self-similarity
            sim[M * N_SZ + gc] = v;
        }
    }
    __syncthreads();

    // ---- Phase 2: top-32 per row; wave w handles rows 2w, 2w+1.
    // Each lane keeps a persistent local max over its stride-32 subset; after
    // each extraction only the owning lane invalidates + rescans its subset.
    for (int rr = 0; rr < 2; ++rr) {
        int r = wave * 2 + rr;
        float* srow = sim + r * N_SZ;

        float lv = NEG_INF;
        int   li = lane;
        for (int i = lane; i < N_SZ; i += 32) {
            float v = srow[i];
            if (v > lv) { lv = v; li = i; }
        }

        for (int k = 0; k < TOPK; ++k) {
            float bv = lv;
            int   bi = li;
#pragma unroll
            for (int off = 16; off >= 1; off >>= 1) {
                float ov = __shfl_xor(bv, off, 32);
                int   oi = __shfl_xor(bi, off, 32);
                if (ov > bv || (ov == bv && oi < bi)) { bv = ov; bi = oi; }
            }
            if (lane == 0) { tval[r * TOPK + k] = bv; tidx[r * TOPK + k] = bi; }
            if (bi == li) {                             // unique owner lane
                srow[bi] = NEG_INF;
                lv = NEG_INF; li = lane;
                for (int i = lane; i < N_SZ; i += 32) {
                    float v = srow[i];
                    if (v > lv) { lv = v; li = i; }
                }
            }
        }
    }
    __syncthreads();

    // ---- Phase 3a: bond_neighbor (sigmoid) + prefetch gathered x rows
    if (tid < ROWS_PER_BLK * TOPK) {                    // 512 entries
        int r = tid >> 5, k = tid & 31;
        float v = tval[tid];
        out_bn[((size_t)(b * N_SZ + nbase + r) << 5) + k] = 1.f / (1.f + __expf(-v));
        __builtin_prefetch(x + ((size_t)(b * N_SZ) + tidx[tid]) * D_SZ, 0, 0);
    }

    // ---- Phase 3b: gather node_neighbor: 16 rows x 32 nbrs x 128 floats
    const float4* x4 = (const float4*)(x + (size_t)b * N_SZ * D_SZ);
    for (int t = tid; t < ROWS_PER_BLK * TOPK * (D_SZ / 4); t += 256) {
        int d4 = t & 31;
        int k  = (t >> 5) & 31;
        int r  = t >> 10;
        int idx = tidx[r * TOPK + k];
        float4 v = x4[(size_t)idx * (D_SZ / 4) + d4];
        float4* dst = (float4*)(out_nn +
            (((size_t)(b * N_SZ + nbase + r) * TOPK + k) * D_SZ));
        dst[d4] = v;
    }
}

extern "C" void kernel_launch(void* const* d_in, const int* in_sizes, int n_in,
                              void* d_out, int out_size, void* d_ws, size_t ws_size,
                              hipStream_t stream) {
    const float* x    = (const float*)d_in[0];   // [8,4096,128]
    const float* ew   = (const float*)d_in[1];   // [1]
    const float* W    = (const float*)d_in[2];   // [10,128]
    const float* bias = (const float*)d_in[3];   // [10]

    float* out_nn = (float*)d_out;                                     // [8,4096,32,128]
    float* out_bn = (float*)d_out + (size_t)B_SZ * N_SZ * TOPK * D_SZ; // [8,4096,32,1]

    _Float16* xt = (_Float16*)d_ws;              // [8,4096,32] f16 = 2 MB

    xt_kernel<<<(B_SZ * N_SZ) / 256, 256, 0, stream>>>(x, W, bias, xt);

    size_t shmem = (size_t)ROWS_PER_BLK * N_SZ * sizeof(float)   // sim slice
                 + (size_t)ROWS_PER_BLK * TOPK * sizeof(float)   // top values
                 + (size_t)ROWS_PER_BLK * TOPK * sizeof(int);    // top indices
    topk_kernel<<<B_SZ * (N_SZ / ROWS_PER_BLK), 256, shmem, stream>>>(
        x, xt, ew, out_nn, out_bn);
}
